// Discriminator3_6786048328063
// MI455X (gfx1250) — matrix operand order repaired and checked by reading, not compiled, not run on hardware
//
#include <hip/hip_runtime.h>
#include <hip/hip_bf16.h>

// ---------------------------------------------------------------------------
// Types for CDNA5 WMMA (wave32, 16x16x32 bf16 -> f32)
// ---------------------------------------------------------------------------
typedef __attribute__((ext_vector_type(16))) __bf16 v16bf;
typedef __attribute__((ext_vector_type(8)))  float  v8f;
typedef __attribute__((ext_vector_type(4)))  float  f32x4;
typedef __attribute__((ext_vector_type(4)))  unsigned u32x4;

union FragBF { v16bf v; unsigned u[8]; };

// Pack two f32 into one dword of two bf16 (truncation) with a single
// v_perm_b32: result = { hi[31:16], lo[31:16] }.
__device__ __forceinline__ unsigned pk_bf16(float lo, float hi) {
  return __builtin_amdgcn_perm(__float_as_uint(hi), __float_as_uint(lo),
                               0x07060302u);
}

// Monotone float<->uint mapping so atomicMax(u32) == float max (handles -inf)
__device__ __forceinline__ unsigned ford(float f) {
  unsigned u = __float_as_uint(f);
  return (u & 0x80000000u) ? ~u : (u | 0x80000000u);
}
__device__ __forceinline__ float forddec(unsigned u) {
  unsigned b = (u & 0x80000000u) ? (u & 0x7FFFFFFFu) : ~u;
  return __uint_as_float(b);
}
#define ORD_NEG_INF 0x007FFFFFu  // ford(-inf)

__device__ __forceinline__ float wave_sum(float v) {
#pragma unroll
  for (int off = 16; off > 0; off >>= 1) v += __shfl_xor(v, off, 32);
  return v;
}

// ---------------------------------------------------------------------------
// A-fragment loader (ISA 7.12.2, 16-bit A 16x32):
// lanes 0-15: row = lane, elems 0..7 = K[kt*32+0..7], elems 8..15 = +16;
// lanes 16-31: row = lane-16, K base +8. Rows clamped (an invalid A row only
// feeds the matching output row, which is never stored). Inputs are
// streamed once -> non-temporal loads (keep L2 for the reused intermediates).
// ---------------------------------------------------------------------------
__device__ __forceinline__ v16bf load_a_bf16(const float* X, int ld, int mbase,
                                             int M, int ktile) {
  int lane = threadIdx.x & 31;
  int half = lane >> 4;
  int row  = mbase + (lane & 15);
  if (row >= M) row = M - 1;
  const f32x4* p =
      (const f32x4*)(X + (size_t)row * ld + ktile * 32 + half * 8);
  f32x4 c0 = __builtin_nontemporal_load(p);
  f32x4 c1 = __builtin_nontemporal_load(p + 1);
  f32x4 c2 = __builtin_nontemporal_load(p + 4);  // +16 floats
  f32x4 c3 = __builtin_nontemporal_load(p + 5);
  FragBF a;
  a.u[0] = pk_bf16(c0.x, c0.y); a.u[1] = pk_bf16(c0.z, c0.w);
  a.u[2] = pk_bf16(c1.x, c1.y); a.u[3] = pk_bf16(c1.z, c1.w);
  a.u[4] = pk_bf16(c2.x, c2.y); a.u[5] = pk_bf16(c2.z, c2.w);
  a.u[6] = pk_bf16(c3.x, c3.y); a.u[7] = pk_bf16(c3.z, c3.w);
  return a.v;
}

// Pre-packed B fragment: 256 dwords per fragment, lane's 8 dwords contiguous.
// 40 KB total -> stays hot in cache; regular-temporal loads.
__device__ __forceinline__ v16bf load_b_packed(const unsigned* wpack, int frag) {
  int lane = threadIdx.x & 31;
  const u32x4* p = (const u32x4*)(wpack + (size_t)frag * 256 + lane * 8);
  u32x4 q0 = p[0];
  u32x4 q1 = p[1];
  FragBF b;
  b.u[0] = q0.x; b.u[1] = q0.y; b.u[2] = q0.z; b.u[3] = q0.w;
  b.u[4] = q1.x; b.u[5] = q1.y; b.u[6] = q1.z; b.u[7] = q1.w;
  return b.v;
}

// C-fragment store: wave-uniform fast path (immediate-offset stores) vs
// guarded tail path. ldo in floats; row stride = ldo*4 bytes (imm offset).
template <int LDO>
__device__ __forceinline__ void store_c(float* out, int mbase, int M,
                                        int outcol, const v8f& c) {
  int lane = threadIdx.x & 31;
  int half = lane >> 4;
  int row0 = mbase + half * 8;
  float* p = out + (size_t)row0 * LDO + outcol;
  if (mbase + 16 <= M) {  // uniform: whole tile in range
#pragma unroll
    for (int r = 0; r < 8; ++r) p[r * LDO] = c[r];
  } else {
#pragma unroll
    for (int r = 0; r < 8; ++r)
      if (row0 + r < M) p[r * LDO] = c[r];
  }
}

// ---------------------------------------------------------------------------
// Kernel P: pack weights [5 x 32 x 128] f32 -> bf16 B-fragments (done once).
// frag id = (w*2 + t)*4 + kt ; w in {q,k,v,skip,e}, t = 16-col tile, kt = K tile.
// B layout per ISA: lane holds column n = t*16 + lane%16;
// lanes 0-15: K = kt*32+0..7,16..23; lanes 16-31: K = kt*32+8..15,24..31.
// ---------------------------------------------------------------------------
__global__ void pack_weights_kernel(const float* __restrict__ Wq,
                                    const float* __restrict__ Wk,
                                    const float* __restrict__ Wv,
                                    const float* __restrict__ Ws,
                                    const float* __restrict__ We,
                                    unsigned* __restrict__ wpack) {
  int t32 = blockIdx.x * blockDim.x + threadIdx.x;
  if (t32 >= 40 * 32) return;
  int frag = t32 >> 5;
  int lane = t32 & 31;
  int w  = frag >> 3;
  int tt = (frag >> 2) & 1;
  int kt = frag & 3;
  const float* Wlist[5] = {Wq, Wk, Wv, Ws, We};
  const float* W = Wlist[w];
  int n     = tt * 16 + (lane & 15);
  int kbase = kt * 32 + (lane >> 4) * 8;
  const float* src = W + (size_t)n * 128 + kbase;
  unsigned* dst = wpack + (size_t)frag * 256 + lane * 8;
#pragma unroll
  for (int j = 0; j < 8; ++j) dst[j] = pk_bf16(src[2 * j + (j & 4) * 2], src[2 * j + (j & 4) * 2 + 1]);
}

// ---------------------------------------------------------------------------
// Kernel 1: fused node GEMM  nodebuf[N][128] = x @ [Wq;Wk;Wv;Wskip]^T + bias
//   cols 0-31=q, 32-63=k, 64-95=v, 96-127=skip. One wave per 16-row tile.
// ---------------------------------------------------------------------------
__global__ __launch_bounds__(128) void node_gemm_kernel(
    const float* __restrict__ x,
    const float* __restrict__ bq, const float* __restrict__ bk,
    const float* __restrict__ bv, const float* __restrict__ bs,
    const unsigned* __restrict__ wpack, float* __restrict__ nodebuf, int Nn) {
  int wave  = blockIdx.x * (blockDim.x >> 5) + (threadIdx.x >> 5);
  int mbase = wave * 16;
  if (mbase >= Nn) return;  // wave-uniform: EXEC stays all-ones for WMMA
  int lane = threadIdx.x & 31;

  v16bf afrag[4];
#pragma unroll
  for (int kt = 0; kt < 4; ++kt) afrag[kt] = load_a_bf16(x, 128, mbase, Nn, kt);

  const float* blist[4] = {bq, bk, bv, bs};

#pragma unroll
  for (int nt = 0; nt < 8; ++nt) {
    const float* bias = blist[nt >> 1];
    float bval = bias[(nt & 1) * 16 + (lane & 15)];
    v8f c;
#pragma unroll
    for (int r = 0; r < 8; ++r) c[r] = bval;
#pragma unroll
    for (int kt = 0; kt < 4; ++kt) {
      v16bf b = load_b_packed(wpack, nt * 4 + kt);
      c = __builtin_amdgcn_wmma_f32_16x16x32_bf16(false, afrag[kt], false, b,
                                                  (short)0, c, false, false);
    }
    store_c<128>(nodebuf, mbase, Nn, nt * 16 + (lane & 15), c);
  }
}

// ---------------------------------------------------------------------------
// Kernel 2: edge GEMM  ebuf[E][32] = edge_attr @ We^T (no bias)
// ---------------------------------------------------------------------------
__global__ __launch_bounds__(128) void edge_gemm_kernel(
    const float* __restrict__ ea, const unsigned* __restrict__ wpack,
    float* __restrict__ ebuf, int Ee) {
  int wave  = blockIdx.x * (blockDim.x >> 5) + (threadIdx.x >> 5);
  int mbase = wave * 16;
  if (mbase >= Ee) return;
  int lane = threadIdx.x & 31;

  v16bf afrag[4];
#pragma unroll
  for (int kt = 0; kt < 4; ++kt) afrag[kt] = load_a_bf16(ea, 128, mbase, Ee, kt);

#pragma unroll
  for (int nt = 0; nt < 2; ++nt) {
    v8f c;
#pragma unroll
    for (int r = 0; r < 8; ++r) c[r] = 0.0f;
#pragma unroll
    for (int kt = 0; kt < 4; ++kt) {
      v16bf b = load_b_packed(wpack, 32 + nt * 4 + kt);  // w=4 (We)
      c = __builtin_amdgcn_wmma_f32_16x16x32_bf16(false, afrag[kt], false, b,
                                                  (short)0, c, false, false);
    }
    store_c<32>(ebuf, mbase, Ee, nt * 16 + (lane & 15), c);
  }
}

// ---------------------------------------------------------------------------
// Kernel 0: init scratch (denom=0, h=0, pooled=0, amax=ordered(-inf))
// ---------------------------------------------------------------------------
__global__ void init_kernel(float* denom, float* h, float* pooled,
                            unsigned* amax, int Nn, int Gg) {
  int t = blockIdx.x * blockDim.x + threadIdx.x;
  if (t < Nn) { denom[t] = 0.0f; amax[t] = ORD_NEG_INF; }
  if (t < Nn * 32) h[t] = 0.0f;
  if (t < Gg * 32) pooled[t] = 0.0f;
}

// ---------------------------------------------------------------------------
// Kernel 3: per-edge attention logit + segment max. One wave per edge,
// lane = channel (coalesced 128B row reads, served from L2-resident buffers).
// ---------------------------------------------------------------------------
__global__ __launch_bounds__(256) void alpha_kernel(
    const int* __restrict__ ei, const float* __restrict__ nodebuf,
    const float* __restrict__ ebuf, float* __restrict__ alpha,
    unsigned* __restrict__ amax, int Ee) {
  int e = blockIdx.x * (blockDim.x >> 5) + (threadIdx.x >> 5);
  if (e >= Ee) return;
  int lane = threadIdx.x & 31;
  int src = ei[e];
  int dst = ei[Ee + e];
  float qv = nodebuf[(size_t)dst * 128 + lane];        // q
  float kv = nodebuf[(size_t)src * 128 + 32 + lane];   // k
  float ev = ebuf[(size_t)e * 32 + lane];
  float s = wave_sum(qv * (kv + ev)) * 0.17677669529663687f;  // 1/sqrt(32)
  if (lane == 0) {
    alpha[e] = s;
    atomicMax(&amax[dst], ford(s));
  }
}

// ---------------------------------------------------------------------------
// Kernel 4: softmax denominator (thread per edge)
// ---------------------------------------------------------------------------
__global__ void denom_kernel(const int* __restrict__ ei,
                             const float* __restrict__ alpha,
                             const unsigned* __restrict__ amax,
                             float* __restrict__ denom, int Ee) {
  int e = blockIdx.x * blockDim.x + threadIdx.x;
  if (e >= Ee) return;
  int dst = ei[Ee + e];
  float m = forddec(amax[dst]);
  if (!isfinite(m)) m = 0.0f;
  atomicAdd(&denom[dst], __expf(alpha[e] - m));
}

// ---------------------------------------------------------------------------
// Kernel 5: weighted message scatter  h[dst] += w * (v[src] + e)
// ---------------------------------------------------------------------------
__global__ __launch_bounds__(256) void scatter_kernel(
    const int* __restrict__ ei, const float* __restrict__ nodebuf,
    const float* __restrict__ ebuf, const float* __restrict__ alpha,
    const unsigned* __restrict__ amax, const float* __restrict__ denom,
    float* __restrict__ h, int Ee) {
  int e = blockIdx.x * (blockDim.x >> 5) + (threadIdx.x >> 5);
  if (e >= Ee) return;
  int lane = threadIdx.x & 31;
  int src = ei[e];
  int dst = ei[Ee + e];
  float m = forddec(amax[dst]);
  if (!isfinite(m)) m = 0.0f;
  float w = __expf(alpha[e] - m) / (denom[dst] + 1e-16f);
  float ve = nodebuf[(size_t)src * 128 + 64 + lane] + ebuf[(size_t)e * 32 + lane];
  atomicAdd(&h[(size_t)dst * 32 + lane], w * ve);
}

// ---------------------------------------------------------------------------
// Kernel 6: add skip connection, pool per graph
// ---------------------------------------------------------------------------
__global__ void pool_kernel(const float* __restrict__ nodebuf,
                            const float* __restrict__ h,
                            const int* __restrict__ batch,
                            float* __restrict__ pooled, int Nn) {
  int t = blockIdx.x * blockDim.x + threadIdx.x;
  if (t >= Nn * 32) return;
  int n = t >> 5, c = t & 31;
  float val = h[t] + nodebuf[(size_t)n * 128 + 96 + c];  // + skip
  atomicAdd(&pooled[(size_t)batch[n] * 32 + c], val);
}

// ---------------------------------------------------------------------------
// Kernel 7: out[g] = tanh(pooled[g]) . Wmlp + bmlp   (wave per graph)
// ---------------------------------------------------------------------------
__global__ __launch_bounds__(256) void head_kernel(
    const float* __restrict__ pooled, const float* __restrict__ Wmlp,
    const float* __restrict__ bmlp, float* __restrict__ out, int Gg) {
  int g = blockIdx.x * (blockDim.x >> 5) + (threadIdx.x >> 5);
  if (g >= Gg) return;
  int lane = threadIdx.x & 31;
  float t = tanhf(pooled[(size_t)g * 32 + lane]);
  float s = wave_sum(t * Wmlp[lane]);
  if (lane == 0) out[g] = s + bmlp[0];
}

// ---------------------------------------------------------------------------
// Host launcher
// ---------------------------------------------------------------------------
extern "C" void kernel_launch(void* const* d_in, const int* in_sizes, int n_in,
                              void* d_out, int out_size, void* d_ws, size_t ws_size,
                              hipStream_t stream) {
  const float* x     = (const float*)d_in[0];
  const int*   ei    = (const int*)d_in[1];
  const float* eattr = (const float*)d_in[2];
  const int*   batch = (const int*)d_in[3];
  const float* Wq = (const float*)d_in[4];  const float* bq = (const float*)d_in[5];
  const float* Wk = (const float*)d_in[6];  const float* bk = (const float*)d_in[7];
  const float* Wv = (const float*)d_in[8];  const float* bv = (const float*)d_in[9];
  const float* We = (const float*)d_in[10];
  const float* Wsk = (const float*)d_in[11]; const float* bsk = (const float*)d_in[12];
  const float* Wmlp = (const float*)d_in[13]; const float* bmlp = (const float*)d_in[14];
  float* out = (float*)d_out;

  const int N = in_sizes[0] / 128;   // 100000
  const int E = in_sizes[2] / 128;   // 640000
  const int G = out_size;            // 1024

  // Workspace layout (all chunks 16B-aligned; ~150 MB total)
  float*    nodebuf = (float*)d_ws;                    // N*128
  float*    ebuf    = nodebuf + (size_t)N * 128;       // E*32
  float*    alpha   = ebuf + (size_t)E * 32;           // E
  unsigned* amax    = (unsigned*)(alpha + E);          // N
  float*    denom   = (float*)(amax + N);              // N
  float*    h       = denom + N;                       // N*32
  float*    pooled  = h + (size_t)N * 32;              // G*32
  unsigned* wpack   = (unsigned*)(pooled + (size_t)G * 32);  // 40*256 dwords

  // P) pack weights into bf16 WMMA-B fragments (one-time, tiny)
  pack_weights_kernel<<<5, 256, 0, stream>>>(Wq, Wk, Wv, Wsk, We, wpack);
  // 0) init accumulators
  init_kernel<<<(N * 32 + 255) / 256, 256, 0, stream>>>(denom, h, pooled, amax, N, G);
  // 1) fused node GEMM (WMMA)
  {
    int waves = (N + 15) / 16;
    node_gemm_kernel<<<(waves + 3) / 4, 128, 0, stream>>>(
        x, bq, bk, bv, bsk, wpack, nodebuf, N);
  }
  // 2) edge GEMM (WMMA)
  {
    int waves = (E + 15) / 16;
    edge_gemm_kernel<<<(waves + 3) / 4, 128, 0, stream>>>(eattr, wpack, ebuf, E);
  }
  // 3) attention logits + segment max
  alpha_kernel<<<(E + 7) / 8, 256, 0, stream>>>(ei, nodebuf, ebuf, alpha, amax, E);
  // 4) softmax denominator
  denom_kernel<<<(E + 255) / 256, 256, 0, stream>>>(ei, alpha, amax, denom, E);
  // 5) weighted message scatter
  scatter_kernel<<<(E + 7) / 8, 256, 0, stream>>>(ei, nodebuf, ebuf, alpha, amax,
                                                  denom, h, E);
  // 6) skip + graph pooling
  pool_kernel<<<(N * 32 + 255) / 256, 256, 0, stream>>>(nodebuf, h, batch, pooled, N);
  // 7) head MLP
  head_kernel<<<(G + 7) / 8, 256, 0, stream>>>(pooled, Wmlp, bmlp, out, G);
}
